// ResidualVQ_90443421319511
// MI455X (gfx1250) — matrix operand order repaired and checked by reading, not compiled, not run on hardware
//
#include <hip/hip_runtime.h>

#define B_   8
#define D_   1024
#define T_   4096
#define Q_   8
#define KC_  1024
#define CD_  32
#define EPS_ 1e-12f
// loss[q] = (1+COMMIT) * mean over (b,c,t) of (z_e - z_q)^2
#define LOSS_SCALE (1.25f / (float)(B_ * CD_ * T_))

typedef __attribute__((ext_vector_type(2)))  float  v2f;
typedef __attribute__((ext_vector_type(8)))  float  v8f;
typedef __attribute__((ext_vector_type(16))) __bf16 v16bf;

// ---------------------------------------------------------------------------
// CDNA5 async copy: 2KB (512 f32) global -> LDS, 32 lanes x 4 x 16B.
// INST_OFFSET is added to both the LDS and global addresses (ISA 8.4.4).
// Tracked on ASYNCcnt; no VGPR return traffic.
// ---------------------------------------------------------------------------
__device__ __forceinline__ void async_stage_2kb(float* buf, const float* src, int lane) {
    uint32_t l = (uint32_t)(uintptr_t)buf + (uint32_t)lane * 16u;   // low 32 bits = LDS offset
    const char* g = (const char*)src + lane * 16;
    asm volatile("global_load_async_to_lds_b128 %0, %1, off"             :: "v"(l), "v"(g) : "memory");
    asm volatile("global_load_async_to_lds_b128 %0, %1, off offset:512"  :: "v"(l), "v"(g) : "memory");
    asm volatile("global_load_async_to_lds_b128 %0, %1, off offset:1024" :: "v"(l), "v"(g) : "memory");
    asm volatile("global_load_async_to_lds_b128 %0, %1, off offset:1536" :: "v"(l), "v"(g) : "memory");
}

// ---------------------------------------------------------------------------
// Pre-pass: L2-normalize codebooks into bf16 rows (WMMA-B friendly: each code
// row is 32 contiguous bf16 = 64B; a lane grabs its 16-element half directly).
// ---------------------------------------------------------------------------
__global__ void rvq_norm_cb(const float* __restrict__ cb, __bf16* __restrict__ cbn) {
    int row = blockIdx.x * blockDim.x + threadIdx.x;   // Q*K rows
    if (row >= Q_ * KC_) return;
    const float* r = cb + (size_t)row * CD_;
    float s = 0.f;
    #pragma unroll
    for (int c = 0; c < CD_; ++c) s += r[c] * r[c];
    float rn = rsqrtf(s + EPS_);
    #pragma unroll
    for (int c = 0; c < CD_; ++c) cbn[(size_t)row * CD_ + c] = (__bf16)(r[c] * rn);
}

__global__ void rvq_zero_loss(float* __restrict__ loss) {
    if (threadIdx.x < Q_) loss[threadIdx.x] = 0.f;
}

// ---------------------------------------------------------------------------
// Fused persistent RVQ kernel: 16 tokens / block, 4 waves, residual in VGPRs.
// ---------------------------------------------------------------------------
__global__ __launch_bounds__(128) void rvq_fused(
    const float* __restrict__ x,    const float* __restrict__ Win,
    const float* __restrict__ b_in, const float* __restrict__ Wout,
    const float* __restrict__ b_out,const float* __restrict__ cb,
    const __bf16* __restrict__ cbn,
    float* __restrict__ out_q, float* __restrict__ out_idx,
    float* __restrict__ out_loss)
{
    __shared__ float lds_part[4][2][8][32];            // per-wave in_proj partial tiles
    __shared__ float lds_ze[16][32];                   // z_e (with bias)
    __shared__ float lds_rn[16];                       // per-token rsqrt norms
    __shared__ float lds_zq[16][32];                   // gathered codebook rows
    __shared__ float lds_bs[4][16];                    // per-wave best score
    __shared__ int   lds_bi[4][16];                    // per-wave best index
    __shared__ float lds_qt[4][16][16];                // out_proj C->A transpose tile
    __shared__ float lds_lr[4];                        // loss partials
    __shared__ __align__(16) float lds_wb[4][2][512];  // async double-buffer: Wout tiles

    const int wave = threadIdx.x >> 5;
    const int lane = threadIdx.x & 31;
    const int lh   = lane & 15;              // M index (token within tile)
    const int hi   = lane >> 4;              // K-half selector in A/B layouts
    const int bb   = blockIdx.x >> 8;        // batch
    const int t0   = (blockIdx.x & 255) << 4;
    const float* xb = x + (size_t)bb * D_ * T_ + t0;
    const int dbase = wave * 256;            // this wave's D-slice

    // Residual in registers, exact WMMA f32 16x16x4 A-layout:
    // k-group jj covers d = dbase+4*jj .. +3; lane holds (K=4jj+2*hi, +1).
    v2f res[64];
    #pragma unroll
    for (int jj = 0; jj < 64; ++jj) {
        int d = dbase + jj * 4 + hi * 2;
        res[jj].x = xb[(size_t)d       * T_ + lh];
        res[jj].y = xb[(size_t)(d + 1) * T_ + lh];
    }

    for (int q = 0; q < Q_; ++q) {
        const float* Wq  = Win  + (size_t)q * CD_ * D_;
        const float* Woq = Wout + (size_t)q * D_ * CD_;
        if (q + 1 < Q_) {                    // warm L2/L0 for next stage
            __builtin_prefetch(Win  + (size_t)(q + 1) * CD_ * D_, 0, 0);
            __builtin_prefetch(Wout + (size_t)(q + 1) * D_ * CD_, 0, 0);
        }

        // ---------- in_proj: z_e[16,32] = res[16,1024] * Win^T ----------
        v8f acc0 = {0,0,0,0,0,0,0,0}, acc1 = {0,0,0,0,0,0,0,0};
        #pragma unroll
        for (int jj = 0; jj < 64; ++jj) {
            int d = dbase + jj * 4 + hi * 2;
            v2f a;  a.x  = res[jj].x;              a.y  = res[jj].y;
            v2f b0; b0.x = Wq[lh * D_ + d];        b0.y = Wq[lh * D_ + d + 1];
            v2f b1; b1.x = Wq[(lh + 16) * D_ + d]; b1.y = Wq[(lh + 16) * D_ + d + 1];
            acc0 = __builtin_amdgcn_wmma_f32_16x16x4_f32(false, a, false, b0, (short)0, acc0, false, false);
            acc1 = __builtin_amdgcn_wmma_f32_16x16x4_f32(false, a, false, b1, (short)0, acc1, false, false);
        }
        #pragma unroll
        for (int v = 0; v < 8; ++v) {
            lds_part[wave][0][v][lane] = acc0[v];
            lds_part[wave][1][v][lane] = acc1[v];
        }
        __syncthreads();
        // cross-wave K-slice reduce + bias (C-layout: elem(t,n): v=t&7, ln=n+16*(t>>3))
        for (int e = threadIdx.x; e < 512; e += 128) {
            int tt = e >> 5, c = e & 31;
            int nt = c >> 4, v = tt & 7, ln = (c & 15) + ((tt >> 3) << 4);
            float s = lds_part[0][nt][v][ln] + lds_part[1][nt][v][ln]
                    + lds_part[2][nt][v][ln] + lds_part[3][nt][v][ln];
            lds_ze[tt][c] = s + b_in[q * CD_ + c];
        }
        __syncthreads();
        if (threadIdx.x < 16) {
            float s = 0.f;
            #pragma unroll
            for (int c = 0; c < CD_; ++c) { float z = lds_ze[threadIdx.x][c]; s += z * z; }
            lds_rn[threadIdx.x] = rsqrtf(s + EPS_);
        }
        __syncthreads();

        // ---------- nearest code: bf16 WMMA dot-scores, argmax ----------
        // Build enc_n as bf16 A operand in contiguous k order (same as B): VGPR v pair -> K = 2v + 16*hi
        float rn = lds_rn[lh];
        v16bf ea;
        #pragma unroll
        for (int v = 0; v < 8; ++v) {
            int kb = 2 * v + hi * 16;
            ea[2 * v]     = (__bf16)(lds_ze[lh][kb]     * rn);
            ea[2 * v + 1] = (__bf16)(lds_ze[lh][kb + 1] * rn);
        }
        float bs[8]; int bi[8];
        #pragma unroll
        for (int v = 0; v < 8; ++v) { bs[v] = -1e30f; bi[v] = 0; }
        #pragma unroll 4
        for (int i = 0; i < 16; ++i) {              // this wave's 256-code slice
            int ct = wave * 16 + i;
            const v16bf* bp = (const v16bf*)(cbn + ((size_t)q * KC_ + (size_t)ct * 16 + lh) * CD_ + hi * 16);
            v16bf eb = *bp;
            v8f sc = {0,0,0,0,0,0,0,0};
            sc = __builtin_amdgcn_wmma_f32_16x16x32_bf16(false, ea, false, eb, (short)0, sc, false, false);
            int code = ct * 16 + lh;
            #pragma unroll
            for (int v = 0; v < 8; ++v) {
                float s = sc[v];
                if (s > bs[v] || (s == bs[v] && code < bi[v])) { bs[v] = s; bi[v] = code; }
            }
        }
        // reduce over the 16 lanes of each half (token m = v + 8*hi), first-index ties
        #pragma unroll
        for (int v = 0; v < 8; ++v) {
            #pragma unroll
            for (int m = 1; m <= 8; m <<= 1) {
                float os = __shfl_xor(bs[v], m);
                int   oi = __shfl_xor(bi[v], m);
                if (os > bs[v] || (os == bs[v] && oi < bi[v])) { bs[v] = os; bi[v] = oi; }
            }
        }
        if (lh == 0) {
            #pragma unroll
            for (int v = 0; v < 8; ++v) { lds_bs[wave][hi * 8 + v] = bs[v]; lds_bi[wave][hi * 8 + v] = bi[v]; }
        }
        __syncthreads();
        if (threadIdx.x < 16) {
            float best = lds_bs[0][threadIdx.x]; int bidx = lds_bi[0][threadIdx.x];
            for (int w = 1; w < 4; ++w)          // wave slices ascend in code id
                if (lds_bs[w][threadIdx.x] > best) { best = lds_bs[w][threadIdx.x]; bidx = lds_bi[w][threadIdx.x]; }
            out_idx[((size_t)q * B_ + bb) * T_ + t0 + threadIdx.x] = (float)bidx;
            const float* crow = cb + ((size_t)q * KC_ + bidx) * CD_;
            #pragma unroll
            for (int c = 0; c < CD_; ++c) lds_zq[threadIdx.x][c] = crow[c];
        }
        __syncthreads();

        // ---------- loss: (1+COMMIT) * mean((z_e - z_q)^2) ----------
        float ll = 0.f;
        for (int e = threadIdx.x; e < 512; e += 128) {
            float d = lds_ze[e >> 5][e & 31] - lds_zq[e >> 5][e & 31];
            ll += d * d;
        }
        #pragma unroll
        for (int m = 1; m < 32; m <<= 1) ll += __shfl_xor(ll, m);
        if (lane == 0) lds_lr[wave] = ll;
        __syncthreads();
        if (threadIdx.x == 0)
            atomicAdd(out_loss + q, (lds_lr[0] + lds_lr[1] + lds_lr[2] + lds_lr[3]) * LOSS_SCALE);

        // ---------- out_proj (+bias) and register-resident residual update ----------
        // Wout tiles (16 rows x 32 cols = 2KB) double-buffered through LDS with
        // CDNA5 async copies so the next tile streams in under the current WMMAs.
        async_stage_2kb(&lds_wb[wave][0][0], Woq + (size_t)dbase * CD_, lane);
        #pragma unroll
        for (int i = 0; i < 16; ++i) {           // 16-wide D tiles in this wave's slice
            int d0 = dbase + i * 16;
            if (i < 15) {
                async_stage_2kb(&lds_wb[wave][(i + 1) & 1][0],
                                Woq + (size_t)(dbase + (i + 1) * 16) * CD_, lane);
                asm volatile("s_wait_asynccnt 4" ::: "memory");   // tile i's 4 copies done
            } else {
                asm volatile("s_wait_asynccnt 0" ::: "memory");
            }
            const float* wt = &lds_wb[wave][i & 1][0];
            v8f acc = {0,0,0,0,0,0,0,0};
            #pragma unroll
            for (int j = 0; j < 8; ++j) {        // K = 32 in groups of 4
                int k = j * 4 + hi * 2;
                v2f a;  a.x  = lds_zq[lh][k];       a.y  = lds_zq[lh][k + 1];
                v2f wv; wv.x = wt[lh * CD_ + k];    wv.y = wt[lh * CD_ + k + 1];
                acc = __builtin_amdgcn_wmma_f32_16x16x4_f32(false, a, false, wv, (short)0, acc, false, false);
            }
            float bo = b_out[q * D_ + d0 + lh];  // N (=d) is fixed per lane
            #pragma unroll
            for (int v = 0; v < 8; ++v)
                lds_qt[wave][v + hi * 8][lh] = acc[v] + bo;   // C-layout -> [token][d_local]
            __syncthreads();
            #pragma unroll
            for (int j2 = 0; j2 < 4; ++j2) {     // read back in A-layout, update residual
                int dl = j2 * 4 + hi * 2;
                res[i * 4 + j2].x -= lds_qt[wave][lh][dl];
                res[i * 4 + j2].y -= lds_qt[wave][lh][dl + 1];
            }
            __syncthreads();
        }
    }

    // quantized_out = x - final residual
    float* ob = out_q + (size_t)bb * D_ * T_ + t0;
    #pragma unroll
    for (int jj = 0; jj < 64; ++jj) {
        int d = dbase + jj * 4 + hi * 2;
        ob[(size_t)d       * T_ + lh] = xb[(size_t)d       * T_ + lh] - res[jj].x;
        ob[(size_t)(d + 1) * T_ + lh] = xb[(size_t)(d + 1) * T_ + lh] - res[jj].y;
    }
}

extern "C" void kernel_launch(void* const* d_in, const int* in_sizes, int n_in,
                              void* d_out, int out_size, void* d_ws, size_t ws_size,
                              hipStream_t stream) {
    const float* x     = (const float*)d_in[0];
    const float* Win   = (const float*)d_in[1];
    const float* b_in  = (const float*)d_in[2];
    const float* Wout  = (const float*)d_in[3];
    const float* b_out = (const float*)d_in[4];
    const float* cb    = (const float*)d_in[5];

    float* out      = (float*)d_out;
    float* out_q    = out;                                    // [B,D,T]
    float* out_idx  = out + (size_t)B_ * D_ * T_;             // [Q,B,T]
    float* out_loss = out_idx + (size_t)Q_ * B_ * T_;         // [Q]

    __bf16* cbn = (__bf16*)d_ws;                              // Q*K*CD bf16 = 512 KB

    rvq_norm_cb<<<(Q_ * KC_ + 255) / 256, 256, 0, stream>>>(cb, cbn);
    rvq_zero_loss<<<1, 32, 0, stream>>>(out_loss);
    rvq_fused<<<(B_ * T_) / 16, 128, 0, stream>>>(x, Win, b_in, Wout, b_out, cb, cbn,
                                                  out_q, out_idx, out_loss);
}
